// Block3d_9715216024052
// MI455X (gfx1250) — matrix operand order repaired and checked by reading, not compiled, not run on hardware
//
#include <hip/hip_runtime.h>
#include <cstdint>

// ---------------------------------------------------------------------------
// Types for CDNA5 WMMA
// ---------------------------------------------------------------------------
typedef __bf16 bf16_t;
typedef bf16_t v16bf __attribute__((ext_vector_type(16)));
typedef float  v8f   __attribute__((ext_vector_type(8)));
typedef int    v4i   __attribute__((ext_vector_type(4)));

union FragAB {
    uint4 q[2];
    v16bf v;
};

// Row-space decomposition (all dims except channel and the mixed axis),
// outermost first. Strides are in the CHANNEL-LAST (L1) layout.
struct RowDecomp {
    int  size[6];
    long stride[6];
};

#define BM 128
#define BN 128
#define BK 32
#define APITCH 40          // halves; 80B rows -> conflict-free 16B chunks
#define BPITCH 40

// ---------------------------------------------------------------------------
// Async copy to LDS (CDNA5) with synchronous fallback
// ---------------------------------------------------------------------------
#define AS3 __attribute__((address_space(3)))

#if defined(__gfx1250__) && __has_builtin(__builtin_amdgcn_global_load_async_to_lds_b128)
#define HAVE_ASYNC 1
#else
#define HAVE_ASYNC 0
#endif

#if HAVE_ASYNC
#if __has_builtin(__builtin_amdgcn_s_wait_asynccnt)
#define WAIT_ASYNC(n) __builtin_amdgcn_s_wait_asynccnt(n)
#else
#define WAIT_ASYNC(n) asm volatile("s_wait_asynccnt " #n ::: "memory")
#endif
__device__ __forceinline__ void cp_b128(const unsigned short* g, unsigned short* l) {
    __builtin_amdgcn_global_load_async_to_lds_b128(
        (v4i*)g, (AS3 v4i*)(unsigned)(uintptr_t)l, 0, 0);
}
#else
#define WAIT_ASYNC(n) do {} while (0)
__device__ __forceinline__ void cp_b128(const unsigned short* g, unsigned short* l) {
    *(uint4*)l = *(const uint4*)g;
}
#endif

// ---------------------------------------------------------------------------
// fp32 -> bf16 (RNE)
// ---------------------------------------------------------------------------
__device__ __forceinline__ unsigned short f2bfbits(float f) {
    unsigned u = __float_as_uint(f);
    unsigned r = u + 0x7FFFu + ((u >> 16) & 1u);
    return (unsigned short)(r >> 16);
}

// Plain convert for weights (layout preserved: [N][K] row-major)
__global__ __launch_bounds__(256) void cvt_f32_bf16(const float* __restrict__ s,
                                                    unsigned short* __restrict__ d,
                                                    long n) {
    long i = ((long)blockIdx.x * blockDim.x + threadIdx.x) * 4;
    if (i >= n) return;
    float4 f = *(const float4*)(s + i);
    ushort4 o;
    o.x = f2bfbits(f.x); o.y = f2bfbits(f.y);
    o.z = f2bfbits(f.z); o.w = f2bfbits(f.w);
    *(ushort4*)(d + i) = o;
}

// ---------------------------------------------------------------------------
// x [b,c,SP] fp32  ->  xb [b,SP,c] bf16   (channel-last, SP = d*h*w*p*q*r)
// 64 (c) x 64 (sp) tiles through LDS.
// ---------------------------------------------------------------------------
__global__ __launch_bounds__(256) void transpose_cvt(const float* __restrict__ x,
                                                     unsigned short* __restrict__ xb) {
    __shared__ unsigned short t[64 * 68];
    const int  b    = blockIdx.x >> 12;            // 4096 sp-tiles per batch
    const int  sp0  = (blockIdx.x & 4095) * 64;
    const long base = (long)b * 16777216L;

    const int tc = threadIdx.x >> 6;   // 0..3
    const int ts = threadIdx.x & 63;
    #pragma unroll
    for (int it = 0; it < 16; ++it) {
        int c = it * 4 + tc;
        t[c * 68 + ts] = f2bfbits(x[base + (long)c * 262144L + sp0 + ts]);
    }
    __syncthreads();
    const int wsp = threadIdx.x >> 4;        // 0..15
    const int wc  = (threadIdx.x & 15) * 4;
    #pragma unroll
    for (int it = 0; it < 4; ++it) {
        int sp = it * 16 + wsp;
        ushort4 o;
        o.x = t[(wc + 0) * 68 + sp];
        o.y = t[(wc + 1) * 68 + sp];
        o.z = t[(wc + 2) * 68 + sp];
        o.w = t[(wc + 3) * 68 + sp];
        *(ushort4*)&xb[base + (long)(sp0 + sp) * 64 + wc] = o;
    }
}

// ---------------------------------------------------------------------------
// Axial GEMM over channel-last bf16 x; fp32 accumulate into channel-last acc.
//   out(m, j=X*64+Y) = sum_{k=s*64+c} xb[rowbase(m) + s*strideA + c] * W[j,k]
//   acc addr = rowbase(m) + X*strideA + Y        (Y stride 1 -> coalesced)
// Double-buffered LDS staging via async b128 copies.
// ---------------------------------------------------------------------------
__global__ __launch_bounds__(256) void axial_gemm(
        const unsigned short* __restrict__ xb,   // bf16, channel-last
        const unsigned short* __restrict__ wb,   // bf16, [N][K] row-major
        const float* __restrict__ bias,          // [N]
        float* __restrict__ acc,                 // fp32, channel-last
        int M, int N, int K, long strideA, RowDecomp rd, int beta)
{
    __shared__ __align__(16) unsigned short As[2][BM * APITCH];
    __shared__ __align__(16) unsigned short Bs[2][BN * BPITCH];
    __shared__ long rowbase[BM];

    const int nTilesN = N / BN;
    const int m0 = (blockIdx.x / nTilesN) * BM;
    const int n0 = (blockIdx.x % nTilesN) * BN;

    const int tid  = threadIdx.x;
    const int lane = tid & 31;
    const int wid  = tid >> 5;
    const int wm   = wid & 3;    // 4 M-waves * 32 rows
    const int wn   = wid >> 2;   // 2 N-waves * 64 cols
    const int mr   = lane & 15;
    const int half = lane >> 4;

    if (tid < BM) {
        long m = m0 + tid;
        long off = 0;
        #pragma unroll
        for (int i = 5; i >= 0; --i) {
            long c = m % rd.size[i];
            m /= rd.size[i];
            off += c * rd.stride[i];
        }
        rowbase[tid] = off;
    }
    __syncthreads();

    // staging map: thread -> (row, 32B half-row), 2 x b128 each for A and B
    const int srow = tid >> 1;
    const int scol = (tid & 1) * 16;          // halves
    const long abase_row = rowbase[srow];

    auto stage = [&](int buf, int k0) {
        const int  s0 = k0 >> 6;              // C == 64
        const int  c0 = k0 & 63;
        const long ka = abase_row + (long)s0 * strideA + c0 + scol;
        const unsigned short* ga = xb + ka;
        const unsigned short* gb = wb + (long)(n0 + srow) * K + k0 + scol;
        unsigned short* la = &As[buf][srow * APITCH + scol];
        unsigned short* lb = &Bs[buf][srow * BPITCH + scol];
        cp_b128(ga,     la);
        cp_b128(ga + 8, la + 8);
        cp_b128(gb,     lb);
        cp_b128(gb + 8, lb + 8);
    };

    v8f Cacc[2][4] = {};

    const int nK = K / BK;
    stage(0, 0);
    for (int t = 0; t < nK; ++t) {
        const int buf = t & 1;
        if (t + 1 < nK) {
            stage(buf ^ 1, (t + 1) * BK);
            WAIT_ASYNC(4);     // our 4 tile-t copies done; next-tile may fly
        } else {
            WAIT_ASYNC(0);
        }
        __syncthreads();

        FragAB a[2];
        #pragma unroll
        for (int ms = 0; ms < 2; ++ms) {
            const unsigned short* p = &As[buf][(wm * 32 + ms * 16 + mr) * APITCH];
            a[ms].q[0] = *(const uint4*)(p + half * 8);        // K 0-7 / 8-15
            a[ms].q[1] = *(const uint4*)(p + 16 + half * 8);   // K 16-23 / 24-31
        }
        #pragma unroll
        for (int ns = 0; ns < 4; ++ns) {
            FragAB b;
            const unsigned short* p =
                &Bs[buf][(wn * 64 + ns * 16 + mr) * BPITCH + half * 8];
            b.q[0] = *(const uint4*)(p);
            b.q[1] = *(const uint4*)(p + 16);
            #pragma unroll
            for (int ms = 0; ms < 2; ++ms) {
                Cacc[ms][ns] = __builtin_amdgcn_wmma_f32_16x16x32_bf16(
                    false, a[ms].v, false, b.v, (short)0, Cacc[ms][ns],
                    false, false);
            }
        }
        __syncthreads();   // release buf for the prefetch at t+2
    }

    // epilogue: coalesced along Y (channel) in channel-last acc
    #pragma unroll
    for (int ns = 0; ns < 4; ++ns) {
        const int  jg     = n0 + wn * 64 + ns * 16 + mr;
        const int  X      = jg >> 6;
        const int  Y      = jg & 63;
        const long colOff = (long)X * strideA + Y;
        const float bj    = bias[jg];
        #pragma unroll
        for (int ms = 0; ms < 2; ++ms) {
            #pragma unroll
            for (int i = 0; i < 8; ++i) {
                int  rrow = wm * 32 + ms * 16 + half * 8 + i;
                long addr = rowbase[rrow] + colOff;
                float v   = Cacc[ms][ns][i] + bj;
                if (beta) acc[addr] += v;
                else      acc[addr]  = v;
            }
        }
    }
}

// ---------------------------------------------------------------------------
// out(L0) = x(L0) + leaky_relu(acc(L1)) with a 64x64 LDS transpose tile
// ---------------------------------------------------------------------------
__global__ __launch_bounds__(256) void epilogue_lrelu_t(const float* __restrict__ x,
                                                        const float* __restrict__ accL1,
                                                        float* __restrict__ outL0) {
    __shared__ float t[64 * 65];
    const int  b    = blockIdx.x >> 12;
    const int  sp0  = (blockIdx.x & 4095) * 64;
    const long base = (long)b * 16777216L;

    const int tc = threadIdx.x >> 6;   // 0..3
    const int ts = threadIdx.x & 63;
    #pragma unroll
    for (int it = 0; it < 16; ++it) {
        int sp = it * 4 + tc;
        t[sp * 65 + ts] = accL1[base + (long)(sp0 + sp) * 64 + ts];
    }
    __syncthreads();
    #pragma unroll
    for (int it = 0; it < 16; ++it) {
        int  c = it * 4 + tc;
        float a  = t[ts * 65 + c];
        float lr = a > 0.f ? a : 0.01f * a;
        long  o  = base + (long)c * 262144L + sp0 + ts;
        outL0[o] = x[o] + lr;
    }
}

// ---------------------------------------------------------------------------
// Host launch
// ---------------------------------------------------------------------------
extern "C" void kernel_launch(void* const* d_in, const int* in_sizes, int n_in,
                              void* d_out, int out_size, void* d_ws, size_t ws_size,
                              hipStream_t stream) {
    (void)in_sizes; (void)n_in; (void)ws_size; (void)out_size;
    const long NTOT = 33554432L;           // 2*64*16^3*4^3
    const float* x = (const float*)d_in[0];

    // channel-last (L1) strides: b 16777216, d 1048576, h 65536, w 4096,
    //                            p 1024, q 256, r 64, c 1
    const int  axS[6]      = {16, 16, 16, 4, 4, 4};
    const long axStride[6] = {1048576, 65536, 4096, 1024, 256, 64};  // d,h,w,p,q,r
    const int  axWidx[6]   = {1, 3, 5, 7, 9, 11};
    const int  axBidx[6]   = {2, 4, 6, 8, 10, 12};
    const RowDecomp rds[6] = {
        {{2,16,16,4,4,4},  {16777216,65536,4096,1024,256,64}},     // axis d
        {{2,16,16,4,4,4},  {16777216,1048576,4096,1024,256,64}},   // axis h
        {{2,16,16,4,4,4},  {16777216,1048576,65536,1024,256,64}},  // axis w
        {{2,16,16,16,4,4}, {16777216,1048576,65536,4096,256,64}},  // axis p
        {{2,16,16,16,4,4}, {16777216,1048576,65536,4096,1024,64}}, // axis q
        {{2,16,16,16,4,4}, {16777216,1048576,65536,4096,1024,256}},// axis r
    };

    // workspace: xb (bf16 L1) | 6 bf16 weights | acc (fp32 L1)
    unsigned short* xb = (unsigned short*)d_ws;
    unsigned short* wbp[6];
    char* p = (char*)d_ws + NTOT * 2;
    for (int a = 0; a < 6; ++a) {
        wbp[a] = (unsigned short*)p;
        long n = (long)(axS[a] * 64) * (long)(axS[a] * 64);
        p += n * 2;
    }
    float* acc = (float*)p;                 // 73.8 MB offset, 16B aligned

    // 1) layout transform + precision convert
    transpose_cvt<<<8192, 256, 0, stream>>>(x, xb);
    for (int a = 0; a < 6; ++a) {
        long n = (long)(axS[a] * 64) * (long)(axS[a] * 64);
        cvt_f32_bf16<<<(int)(n / 1024), 256, 0, stream>>>(
            (const float*)d_in[axWidx[a]], wbp[a], n);
    }

    // 2) six axial GEMMs accumulating into acc (channel-last fp32)
    for (int a = 0; a < 6; ++a) {
        const int N = axS[a] * 64;
        const int K = N;
        const int M = (int)(NTOT / N);
        const int grid = (M / BM) * (N / BN);
        axial_gemm<<<grid, 256, 0, stream>>>(
            xb, wbp[a], (const float*)d_in[axBidx[a]], acc,
            M, N, K, axStride[a], rds[a], /*beta=*/(a != 0));
    }

    // 3) transpose back + residual + leaky relu
    epilogue_lrelu_t<<<8192, 256, 0, stream>>>(x, acc, (float*)d_out);
}
